// MultiHeadAttention_44676249813603
// MI455X (gfx1250) — matrix operand order repaired and checked by reading, not compiled, not run on hardware
//
#include <hip/hip_runtime.h>
#include <cstdint>

// ---------------------------------------------------------------------------
// Multi-head attention for MI455X (gfx1250), bf16 WMMA pipeline with
// async-to-LDS staging (ASYNCcnt) and Tensor Data Mover staging (TENSORcnt).
// B=8, S=1024, D_MODEL=512, H=8, U=64.
// ---------------------------------------------------------------------------

typedef unsigned short u16;
typedef __bf16 bf16;
typedef __attribute__((ext_vector_type(4)))  unsigned u32x4;
typedef __attribute__((ext_vector_type(8)))  unsigned u32x8;
typedef __attribute__((ext_vector_type(8)))  u16   u16x8;
typedef __attribute__((ext_vector_type(16))) u16   u16x16;
typedef __attribute__((ext_vector_type(16))) bf16  bf16x16;
typedef __attribute__((ext_vector_type(8)))  float f32x8;

#define BATCHN 8
#define SEQL   1024
#define DMODEL 512
#define HEADS  8
#define UDIM   64
#define ROWS   (BATCHN * SEQL)          // 8192
#define CONCAT (HEADS * DMODEL)         // 4096

__device__ __forceinline__ u16 f32_to_bf16(float f) {
  unsigned x = __builtin_bit_cast(unsigned, f);
  x += 0x7FFFu + ((x >> 16) & 1u);      // round-to-nearest-even
  return (u16)(x >> 16);
}

__device__ __forceinline__ bf16x16 as_bf16x16(u16x8 lo, u16x8 hi) {
  union { u16x16 u; bf16x16 b; } cvt;
#pragma unroll
  for (int i = 0; i < 8; i++) { cvt.u[i] = lo[i]; cvt.u[8 + i] = hi[i]; }
  return cvt.b;
}

__device__ __forceinline__ f32x8 zero8() {
  f32x8 z;
#pragma unroll
  for (int i = 0; i < 8; i++) z[i] = 0.0f;
  return z;
}

__device__ __forceinline__ f32x8 wmma_bf16(bf16x16 a, bf16x16 b, f32x8 c) {
  return __builtin_amdgcn_wmma_f32_16x16x32_bf16(false, a, false, b,
                                                 (short)0, c, false, false);
}

// Async copy of one 16-byte chunk: global -> LDS (GV mode, per-lane).
__device__ __forceinline__ void async_b128_to_lds(const u16* gptr, u16* lptr) {
  unsigned lds = (unsigned)(uintptr_t)lptr;               // low 32 bits = LDS offset
  unsigned long long ga = (unsigned long long)(uintptr_t)gptr;
  asm volatile("global_load_async_to_lds_b128 %0, %1, off"
               :: "v"(lds), "v"(ga) : "memory");
}

__device__ __forceinline__ void wait_asynccnt0() {
  asm volatile("s_wait_asynccnt 0" ::: "memory");
}

// ---------------------------------------------------------------------------
// Elementwise casts
// ---------------------------------------------------------------------------
__global__ __launch_bounds__(256) void cast_f32_bf16_kernel(
    const float* __restrict__ src, u16* __restrict__ dst, int n) {
  int idx = blockIdx.x * 256 + threadIdx.x;
  if (idx < n) dst[idx] = f32_to_bf16(src[idx]);
}

// Batched transpose+cast: src [batch][R][C] f32 -> dst [batch][C][R] bf16
__global__ __launch_bounds__(256) void transpose_cast_kernel(
    const float* __restrict__ src, u16* __restrict__ dst,
    int R, int C, int total) {
  int idx = blockIdx.x * 256 + threadIdx.x;
  if (idx >= total) return;
  int c  = idx % C;
  int r  = (idx / C) % R;
  int bb = idx / (R * C);
  dst[(size_t)bb * R * C + (size_t)c * R + r] = f32_to_bf16(src[idx]);
}

// ---------------------------------------------------------------------------
// Generic bf16 GEMM: C[M,N] = A[M,K] * B[K,N] with B supplied TRANSPOSED
// (BT[N,K] row-major). The 64xKC BT panel is staged in LDS with async loads
// and shared by all 8 waves; A fragments stream from global (per-wave rows).
//   OUT_MODE 0: bf16 row-major store   (ldc = N)
//   OUT_MODE 1: bf16 transposed store  (ldc = M; C[n*M + m])
//   OUT_MODE 2: f32  row-major store   (ldc = N)
// Grids must tile exactly: M%128==0, N%64==0, K%128==0.
// ---------------------------------------------------------------------------
template <int OUT_MODE>
__global__ __launch_bounds__(256) void gemm_bf16_kernel(
    const u16* __restrict__ A, const u16* __restrict__ BT,
    void* __restrict__ Cv, int M, int N, int K,
    int lda, int ldbt, int ldc) {
  __shared__ u16 Blds[64 * 128];       // 16 KB: [64 cols][128 K] bf16

  const int tid  = threadIdx.x;
  const int lane = tid & 31;
  const int wave = tid >> 5;
  const int half = lane >> 4;
  const int l16  = lane & 15;
  const int m0 = blockIdx.y * 128 + wave * 16;
  const int n0 = blockIdx.x * 64;

  f32x8 acc[4];
#pragma unroll
  for (int t = 0; t < 4; t++) acc[t] = zero8();

  const u16* arow = A + (size_t)(m0 + l16) * lda;

  for (int kc0 = 0; kc0 < K; kc0 += 128) {
    __syncthreads();                    // previous panel fully consumed
    // Stage BT panel: 64 cols x 128 K = 1024 x 16B chunks, 4 per thread.
#pragma unroll
    for (int j = 0; j < 4; j++) {
      int c  = tid + 256 * j;
      int lc = c >> 4;                  // col 0..63
      int ch = c & 15;                  // 16B chunk within 256B row
      async_b128_to_lds(BT + (size_t)(n0 + lc) * ldbt + kc0 + ch * 8,
                        Blds + lc * 128 + ch * 8);
    }
    wait_asynccnt0();
    __syncthreads();                    // panel visible to all waves

#pragma unroll
    for (int kk = 0; kk < 128; kk += 32) {
      int k0 = kc0 + kk;
      // A fragment 16x32: lane holds row (lane&15); K = 8h..+7, 16+8h..+7
      u16x8 a_lo = *(const u16x8*)(arow + k0 + 8 * half);
      u16x8 a_hi = *(const u16x8*)(arow + k0 + 16 + 8 * half);
      bf16x16 afrag = as_bf16x16(a_lo, a_hi);
#pragma unroll
      for (int t = 0; t < 4; t++) {
        // B fragment 32x16 from LDS: lane holds col; K = 8h..+7, 16+8h..+7
        const u16* brow = Blds + (t * 16 + l16) * 128 + kk + 8 * half;
        u16x8 b_lo = *(const u16x8*)(brow);
        u16x8 b_hi = *(const u16x8*)(brow + 16);
        acc[t] = wmma_bf16(afrag, as_bf16x16(b_lo, b_hi), acc[t]);
      }
    }
  }

#pragma unroll
  for (int t = 0; t < 4; t++) {
    int col = n0 + t * 16 + l16;
#pragma unroll
    for (int i = 0; i < 8; i++) {
      int row = m0 + half * 8 + i;
      float v = acc[t][i];
      if (OUT_MODE == 2)
        ((float*)Cv)[(size_t)row * ldc + col] = v;
      else if (OUT_MODE == 1)
        ((u16*)Cv)[(size_t)col * ldc + row] = f32_to_bf16(v);
      else
        ((u16*)Cv)[(size_t)row * ldc + col] = f32_to_bf16(v);
    }
  }
}

// ---------------------------------------------------------------------------
// Flash-style attention. Grid: (S/64, B*H). Block = 256 = 8 waves.
// Per key block of 64:
//   * wave 0 DMAs the 64KB v-block [512 feat x 64 keys] into LDS via the
//     Tensor Data Mover (one 2D D# descriptor, TENSORcnt)
//   * all waves async-load the 8KB k-block [64 keys x 64 U] into LDS
//   * scores, online softmax, P re-layout through LDS, PV from LDS v-block
//   qbuf/kbuf : [H][8192][64]  bf16 ; vT : [H][512][8192] bf16
//   attn      : [8192][4096]   bf16 (concat layout, col = h*512 + f)
// ---------------------------------------------------------------------------
__global__ __launch_bounds__(256) void attention_kernel(
    const u16* __restrict__ qbuf, const u16* __restrict__ kbuf,
    const u16* __restrict__ vT, u16* __restrict__ attn) {
  __shared__ u16 Vlds[512 * 64];      // 64 KB: [feature][key] bf16
  __shared__ u16 Klds[64 * 64];       //  8 KB: [key][U] bf16
  __shared__ u16 Plds[8 * 16 * 64];   // 16 KB: per-wave 16x64 P tile

  const int tid  = threadIdx.x;
  const int lane = tid & 31;
  const int wave = tid >> 5;
  const int half = lane >> 4;
  const int l16  = lane & 15;
  const int wm   = wave & 3;
  const int wn   = wave >> 2;

  const int bh = blockIdx.y;          // b*H + h
  const int b  = bh >> 3;
  const int h  = bh & 7;
  const int s0 = blockIdx.x * 64;

  const u16* qh = qbuf + ((size_t)h * ROWS + (size_t)b * SEQL) * UDIM;
  const u16* kh = kbuf + ((size_t)h * ROWS + (size_t)b * SEQL) * UDIM;
  const u16* vh = vT   + (size_t)h * DMODEL * ROWS + (size_t)b * SEQL;

  // Loop-invariant q A-fragments (16 rows x 64 U, two K-steps of 32)
  bf16x16 aq[2];
  {
    const u16* qrow = qh + (size_t)(s0 + wm * 16 + l16) * UDIM;
#pragma unroll
    for (int kc = 0; kc < 2; kc++) {
      u16x8 lo = *(const u16x8*)(qrow + kc * 32 + 8 * half);
      u16x8 hi = *(const u16x8*)(qrow + kc * 32 + 16 + 8 * half);
      aq[kc] = as_bf16x16(lo, hi);
    }
  }

  f32x8 acc[16];
#pragma unroll
  for (int ft = 0; ft < 16; ft++) acc[ft] = zero8();
  float mrow[8], lrow[8];
#pragma unroll
  for (int i = 0; i < 8; i++) { mrow[i] = -1e30f; lrow[i] = 0.0f; }

  const float scale = 0.04419417382415922f;   // 1/sqrt(512)
  const float LOG2E = 1.4426950408889634f;
  u16* pw = Plds + wave * (16 * 64);

  for (int kb0 = 0; kb0 < SEQL; kb0 += 64) {
    __syncthreads();                  // LDS buffers free for reuse

    if (wave == 0) {
      // ---- TDM: DMA v-block [512 rows x 64 keys, row stride 8192] -> LDS ----
      unsigned long long ga = (unsigned long long)(uintptr_t)(vh + kb0);
      unsigned ldsa = (unsigned)(uintptr_t)Vlds;
      u32x4 g0;
      g0[0] = 1u;                                   // count=1, user descriptor
      g0[1] = ldsa;                                 // lds_addr
      g0[2] = (unsigned)ga;                         // global_addr[31:0]
      g0[3] = ((unsigned)(ga >> 32) & 0x01FFFFFFu)  // global_addr[56:32]
              | 0x80000000u;                        // type=2 ("image")
      u32x8 g1;
      g1[0] = 0x00010000u;          // workgroup_mask=0, data_size=1 (2B)
      g1[1] = (1024u << 16);        // tensor_dim0[15:0]=1024
      g1[2] = (512u << 16);         // tensor_dim0[31:16]=0 | tensor_dim1[15:0]=512
      g1[3] = (64u << 16);          // tensor_dim1[31:16]=0 | tile_dim0=64
      g1[4] = 512u;                 // tile_dim1=512, tile_dim2=0
      g1[5] = 8192u;                // tensor_dim0_stride[31:0]=8192 elements
      g1[6] = 0u;                   // stride hi / tensor_dim1_stride lo
      g1[7] = 0u;
      asm volatile("tensor_load_to_lds %0, %1" :: "s"(g0), "s"(g1) : "memory");
    }

    // ---- async-load k-block [64 keys x 64 U] -> LDS (2 chunks/thread) ----
#pragma unroll
    for (int j = 0; j < 2; j++) {
      int c   = tid + 256 * j;      // 512 x 16B chunks
      int row = c >> 3;
      int ch  = c & 7;
      async_b128_to_lds(kh + (size_t)(kb0 + row) * UDIM + ch * 8,
                        Klds + row * UDIM + ch * 8);
    }
    wait_asynccnt0();
    if (wave == 0) asm volatile("s_wait_tensorcnt 0" ::: "memory");
    __syncthreads();                  // staged data visible to all waves

    // ---- scores S = q * k^T from LDS k-block ----
    f32x8 st[4];
#pragma unroll
    for (int t = 0; t < 4; t++) st[t] = zero8();
#pragma unroll
    for (int t = 0; t < 4; t++) {
      const u16* krow = Klds + (t * 16 + l16) * UDIM;
#pragma unroll
      for (int kc = 0; kc < 2; kc++) {
        u16x8 lo = *(const u16x8*)(krow + kc * 32 + 8 * half);
        u16x8 hi = *(const u16x8*)(krow + kc * 32 + 16 + 8 * half);
        st[t] = wmma_bf16(aq[kc], as_bf16x16(lo, hi), st[t]);
      }
    }

    // ---- online softmax (rows live across 16-lane halves) ----
#pragma unroll
    for (int i = 0; i < 8; i++) {
      float r = fmaxf(fmaxf(st[0][i], st[1][i]), fmaxf(st[2][i], st[3][i]));
      r = fmaxf(r, __shfl_xor(r, 1));
      r = fmaxf(r, __shfl_xor(r, 2));
      r = fmaxf(r, __shfl_xor(r, 4));
      r = fmaxf(r, __shfl_xor(r, 8));
      float mnew = fmaxf(mrow[i], r * scale);
      float corr = __builtin_amdgcn_exp2f((mrow[i] - mnew) * LOG2E);
      mrow[i] = mnew;
      float rs = 0.0f;
#pragma unroll
      for (int t = 0; t < 4; t++) {
        float p = __builtin_amdgcn_exp2f((st[t][i] * scale - mnew) * LOG2E);
        rs += p;
        pw[(half * 8 + i) * 64 + t * 16 + l16] = f32_to_bf16(p);
      }
      rs += __shfl_xor(rs, 1);
      rs += __shfl_xor(rs, 2);
      rs += __shfl_xor(rs, 4);
      rs += __shfl_xor(rs, 8);
      lrow[i] = lrow[i] * corr + rs;
#pragma unroll
      for (int ft = 0; ft < 16; ft++) acc[ft][i] *= corr;
    }

    // ---- P as A-fragments from LDS ----
    bf16x16 ap[2];
#pragma unroll
    for (int kc = 0; kc < 2; kc++) {
      const u16* prow = pw + l16 * 64 + kc * 32;
      u16x8 lo = *(const u16x8*)(prow + 8 * half);
      u16x8 hi = *(const u16x8*)(prow + 16 + 8 * half);
      ap[kc] = as_bf16x16(lo, hi);
    }

    // ---- PV from LDS v-block ----
#pragma unroll
    for (int ft = 0; ft < 16; ft++) {
      const u16* vrow = Vlds + (wn * 256 + ft * 16 + l16) * 64;
#pragma unroll
      for (int kc = 0; kc < 2; kc++) {
        u16x8 lo = *(const u16x8*)(vrow + kc * 32 + 8 * half);
        u16x8 hi = *(const u16x8*)(vrow + kc * 32 + 16 + 8 * half);
        acc[ft] = wmma_bf16(ap[kc], as_bf16x16(lo, hi), acc[ft]);
      }
    }
  }

  // ---- epilogue: normalize and store to concat buffer ----
  float rinv[8];
#pragma unroll
  for (int i = 0; i < 8; i++) rinv[i] = 1.0f / lrow[i];
  const size_t rowbase = (size_t)b * SEQL + s0 + wm * 16 + half * 8;
#pragma unroll
  for (int ft = 0; ft < 16; ft++) {
    int col = h * DMODEL + wn * 256 + ft * 16 + l16;
#pragma unroll
    for (int i = 0; i < 8; i++)
      attn[(rowbase + i) * CONCAT + col] = f32_to_bf16(acc[ft][i] * rinv[i]);
  }
}

// ---------------------------------------------------------------------------
// Host-side orchestration
// ---------------------------------------------------------------------------
extern "C" void kernel_launch(void* const* d_in, const int* in_sizes, int n_in,
                              void* d_out, int out_size, void* d_ws, size_t ws_size,
                              hipStream_t stream) {
  (void)in_sizes; (void)n_in; (void)out_size; (void)ws_size;
  const float* Q  = (const float*)d_in[0];
  const float* K  = (const float*)d_in[1];
  const float* V  = (const float*)d_in[2];
  const float* Wq = (const float*)d_in[3];
  const float* Wk = (const float*)d_in[4];
  const float* Wv = (const float*)d_in[5];
  const float* Wl = (const float*)d_in[6];
  float* out = (float*)d_out;

  char* ws = (char*)d_ws;
  size_t off = 0;
  auto alloc = [&](size_t bytes) -> void* {
    void* p = ws + off;
    off += (bytes + 255) & ~(size_t)255;
    return p;
  };
  u16* Qb   = (u16*)alloc((size_t)ROWS * DMODEL * 2);          // 8 MB
  u16* Kb   = (u16*)alloc((size_t)ROWS * DMODEL * 2);          // 8 MB
  u16* Vb   = (u16*)alloc((size_t)ROWS * DMODEL * 2);          // 8 MB
  u16* WqT  = (u16*)alloc((size_t)HEADS * UDIM * DMODEL * 2);  // 512 KB
  u16* WkT  = (u16*)alloc((size_t)HEADS * UDIM * DMODEL * 2);  // 512 KB
  u16* WvT  = (u16*)alloc((size_t)HEADS * DMODEL * DMODEL * 2);// 4 MB
  u16* WlT  = (u16*)alloc((size_t)DMODEL * CONCAT * 2);        // 4 MB
  u16* qbuf = (u16*)alloc((size_t)HEADS * ROWS * UDIM * 2);    // 8 MB
  u16* kbuf = (u16*)alloc((size_t)HEADS * ROWS * UDIM * 2);    // 8 MB
  u16* vT   = (u16*)alloc((size_t)HEADS * DMODEL * ROWS * 2);  // 64 MB
  u16* attn = (u16*)alloc((size_t)ROWS * CONCAT * 2);          // 64 MB

  const int nQKV = ROWS * DMODEL;  // 4,194,304
  cast_f32_bf16_kernel<<<nQKV / 256, 256, 0, stream>>>(Q, Qb, nQKV);
  cast_f32_bf16_kernel<<<nQKV / 256, 256, 0, stream>>>(K, Kb, nQKV);
  cast_f32_bf16_kernel<<<nQKV / 256, 256, 0, stream>>>(V, Vb, nQKV);

  {
    int tot = HEADS * DMODEL * UDIM;   // 262,144
    transpose_cast_kernel<<<tot / 256, 256, 0, stream>>>(Wq, WqT, DMODEL, UDIM, tot);
    transpose_cast_kernel<<<tot / 256, 256, 0, stream>>>(Wk, WkT, DMODEL, UDIM, tot);
  }
  {
    int tot = HEADS * DMODEL * DMODEL; // 2,097,152
    transpose_cast_kernel<<<tot / 256, 256, 0, stream>>>(Wv, WvT, DMODEL, DMODEL, tot);
  }
  {
    int tot = CONCAT * DMODEL;         // 2,097,152
    transpose_cast_kernel<<<tot / 256, 256, 0, stream>>>(Wl, WlT, CONCAT, DMODEL, tot);
  }

  // Per-head projections: q,k [8192x512]x[512x64]; v (transposed out) x[512x512]
  for (int h = 0; h < HEADS; h++) {
    gemm_bf16_kernel<0><<<dim3(UDIM / 64, ROWS / 128), 256, 0, stream>>>(
        Qb, WqT + (size_t)h * UDIM * DMODEL, qbuf + (size_t)h * ROWS * UDIM,
        ROWS, UDIM, DMODEL, DMODEL, DMODEL, UDIM);
    gemm_bf16_kernel<0><<<dim3(UDIM / 64, ROWS / 128), 256, 0, stream>>>(
        Kb, WkT + (size_t)h * UDIM * DMODEL, kbuf + (size_t)h * ROWS * UDIM,
        ROWS, UDIM, DMODEL, DMODEL, DMODEL, UDIM);
    gemm_bf16_kernel<1><<<dim3(DMODEL / 64, ROWS / 128), 256, 0, stream>>>(
        Vb, WvT + (size_t)h * DMODEL * DMODEL, vT + (size_t)h * DMODEL * ROWS,
        ROWS, DMODEL, DMODEL, DMODEL, DMODEL, ROWS /*transposed ldc*/);
  }

  // Flash attention -> concat buffer
  attention_kernel<<<dim3(SEQL / 64, BATCHN * HEADS), 256, 0, stream>>>(
      qbuf, kbuf, vT, attn);

  // Final projection: [8192x4096] x [4096x512] -> f32 out
  gemm_bf16_kernel<2><<<dim3(DMODEL / 64, ROWS / 128), 256, 0, stream>>>(
      attn, WlT, out, ROWS, DMODEL, CONCAT, CONCAT, CONCAT, DMODEL);
}
